// Decoder_16612933500998
// MI455X (gfx1250) — compile-verified, matrix operand order checked by reading, not run on hardware
//
#include <hip/hip_runtime.h>

#define B_TOK 1024
#define DIN   512
#define HID   2048
#define OUTD  8192
#define NEXP  5

typedef __attribute__((ext_vector_type(16))) __bf16 v16bf;
typedef __attribute__((ext_vector_type(8)))  float  v8f;

// ---------------------------------------------------------------- routing ---
// Single wave, stable compaction of token ids by expert.
__global__ __launch_bounds__(32) void route_kernel(const float* __restrict__ x,
                                                   int* __restrict__ segoff,  // 6 ints
                                                   int* __restrict__ idxs)    // 1024 ints
{
    const int lane = threadIdx.x;
    const unsigned long long ltmask = (1ull << lane) - 1ull;

    int cnt[NEXP] = {0, 0, 0, 0, 0};
    for (int c = 0; c < B_TOK / 32; ++c) {
        const int t = c * 32 + lane;
        const bool c1 = ((int)x[t * DIN + DIN - 1]) == 0;
        const bool c2 = ((int)x[t * DIN + DIN - 2]) == 0;
        const bool c3 = ((int)x[t * DIN + DIN - 3]) == 0;
        const int e = c1 ? (c2 ? 0 : (c3 ? 3 : 4)) : (c2 ? 1 : 2);
#pragma unroll
        for (int E = 0; E < NEXP; ++E)
            cnt[E] += __popcll(__ballot(e == E));
    }
    int off[NEXP + 1];
    off[0] = 0;
#pragma unroll
    for (int E = 0; E < NEXP; ++E) off[E + 1] = off[E] + cnt[E];
    if (lane < NEXP + 1) segoff[lane] = off[lane];

    int run[NEXP];
#pragma unroll
    for (int E = 0; E < NEXP; ++E) run[E] = off[E];

    for (int c = 0; c < B_TOK / 32; ++c) {
        const int t = c * 32 + lane;
        const bool c1 = ((int)x[t * DIN + DIN - 1]) == 0;
        const bool c2 = ((int)x[t * DIN + DIN - 2]) == 0;
        const bool c3 = ((int)x[t * DIN + DIN - 3]) == 0;
        const int e = c1 ? (c2 ? 0 : (c3 ? 3 : 4)) : (c2 ? 1 : 2);
#pragma unroll
        for (int E = 0; E < NEXP; ++E) {
            const unsigned long long m = __ballot(e == E);
            if (e == E) idxs[run[E] + (int)__popcll(m & ltmask)] = t;
            run[E] += (int)__popcll(m);
        }
    }
}

// ------------------------------------------------------------- WMMA helper ---
// Load one 16xK(=32) bf16 fragment from an LDS tile whose rows (length 32,
// stride 36 halves) hold either A rows (M) or transposed-B rows (N).
// Layout per CDNA5 ISA 7.12.2: lanes 0-15 rows 0-15 (K 0-7 / 16-23 halves),
// lanes 16-31 same rows (K 8-15 / 24-31).
__device__ __forceinline__ v16bf load_frag(const __bf16* __restrict__ base, int lane)
{
    const __bf16* row = base + (lane & 15) * 36;
    const int kh = (lane >> 4) * 8;
    v16bf f;
#pragma unroll
    for (int j = 0; j < 4; ++j) {
        f[2 * j]     = row[kh + 2 * j];
        f[2 * j + 1] = row[kh + 2 * j + 1];
    }
#pragma unroll
    for (int j = 0; j < 4; ++j) {
        f[8 + 2 * j]     = row[16 + kh + 2 * j];
        f[8 + 2 * j + 1] = row[16 + kh + 2 * j + 1];
    }
    return f;
}

// ---------------------------------------------------------- grouped GEMM ----
// PASS2=false: H[off+g] = relu(x[idx[off+g]] @ W1[e] + b1[e])   (K=512,  N=2048)
// PASS2=true : out[idx[off+g]] = H[off+g] @ W2[e] + b2[e]       (K=2048, N=8192)
// Block tile 256(M) x 128(N), K-step 32, 8 waves as 4(M) x 2(N), wave tile 64x64.
// Pass 2 stages the (already-bf16) A tile with global_load_async_to_lds_b128,
// overlapping the copy with the B-tile f32->bf16 convert/transpose.
template <bool PASS2>
__global__ __launch_bounds__(256) void moe_gemm(const float*  __restrict__ X,
                                                const float*  __restrict__ W,
                                                const float*  __restrict__ bias,
                                                const __bf16* __restrict__ Hin,
                                                __bf16*       __restrict__ Hout,
                                                float*        __restrict__ O,
                                                const int*    __restrict__ segoff,
                                                const int*    __restrict__ idxs)
{
    constexpr int BM = 256, BN = 128, KS = 32, LDT = 36;
    constexpr int KDIM = PASS2 ? HID : DIN;
    constexpr int NDIM = PASS2 ? OUTD : HID;

    __shared__ __bf16 Atile[BM * LDT];   // [m][k], padded rows
    __shared__ __bf16 Btile[BN * LDT];   // transposed: [n][k], padded rows

    const int e    = blockIdx.z;
    const int off  = segoff[e];
    const int n_e  = segoff[e + 1] - off;
    const int m0   = blockIdx.y * BM;
    if (m0 >= n_e) return;               // uniform early-exit for empty tiles
    const int n0   = blockIdx.x * BN;

    const int tid  = threadIdx.x;
    const int lane = tid & 31;
    const int wave = tid >> 5;
    const int wm   = (wave & 3) * 64;    // wave M offset in block tile
    const int wn   = (wave >> 2) * 64;   // wave N offset in block tile

    const float* Wb = W + (size_t)e * KDIM * NDIM;

    // A-staging source for this thread's tile row (rows past n_e duplicate the
    // last valid row; their outputs are masked off in the epilogue).
    const int g  = m0 + tid;
    const int gc = (g < n_e) ? g : (n_e - 1);
    const float*  arow_f = nullptr;
    const __bf16* arow_h = nullptr;
    if constexpr (!PASS2) {
        arow_f = X + (size_t)idxs[off + gc] * DIN;
    } else {
        arow_h = Hin + (size_t)(off + gc) * HID;
    }
    // Raw LDS byte offset of this thread's A-tile row (low 32 bits of the flat
    // shared-aperture address are the LDS offset on CDNA5).
    const unsigned lds_a = (unsigned)(size_t)&Atile[tid * LDT];

    v8f acc[4][4];
#pragma unroll
    for (int mi = 0; mi < 4; ++mi)
#pragma unroll
        for (int ni = 0; ni < 4; ++ni)
#pragma unroll
            for (int r = 0; r < 8; ++r) acc[mi][ni][r] = 0.0f;

    for (int k0 = 0; k0 < KDIM; k0 += KS) {
        // ---- stage A tile: thread tid owns tile row tid (32 halves = 64 B) ----
        if constexpr (!PASS2) {
            const float4* src = (const float4*)(arow_f + k0);
            __bf16* dst = &Atile[tid * LDT];
#pragma unroll
            for (int q = 0; q < 8; ++q) {
                const float4 v = src[q];
                dst[q * 4 + 0] = (__bf16)v.x;
                dst[q * 4 + 1] = (__bf16)v.y;
                dst[q * 4 + 2] = (__bf16)v.z;
                dst[q * 4 + 3] = (__bf16)v.w;
            }
        } else {
            // Async copy global(bf16) -> LDS: 4 x 16 B per thread. The shared
            // INST_OFFSET advances both the global and LDS addresses, and the
            // 64 B payload is contiguous on both sides (pad lies between rows).
            const __bf16* gsrc = arow_h + k0;
            asm volatile("global_load_async_to_lds_b128 %0, %1, off"
                         :: "v"(lds_a), "v"(gsrc) : "memory");
            asm volatile("global_load_async_to_lds_b128 %0, %1, off offset:16"
                         :: "v"(lds_a), "v"(gsrc) : "memory");
            asm volatile("global_load_async_to_lds_b128 %0, %1, off offset:32"
                         :: "v"(lds_a), "v"(gsrc) : "memory");
            asm volatile("global_load_async_to_lds_b128 %0, %1, off offset:48"
                         :: "v"(lds_a), "v"(gsrc) : "memory");
        }
        // ---- stage B tile transposed: 32(k) x 128(n) -> Btile[n][k] ----
        {
            const int kk = tid >> 3;               // 0..31
            const int nq = (tid & 7) * 16;         // 0,16,...,112
            const float* srow = Wb + (size_t)(k0 + kk) * NDIM + n0 + nq;
#pragma unroll
            for (int q = 0; q < 4; ++q) {
                const float4 v = *(const float4*)(srow + q * 4);
                Btile[(nq + q * 4 + 0) * LDT + kk] = (__bf16)v.x;
                Btile[(nq + q * 4 + 1) * LDT + kk] = (__bf16)v.y;
                Btile[(nq + q * 4 + 2) * LDT + kk] = (__bf16)v.z;
                Btile[(nq + q * 4 + 3) * LDT + kk] = (__bf16)v.w;
            }
        }
        if constexpr (PASS2) {
            asm volatile("s_wait_asynccnt 0x0" ::: "memory");
        }
        __syncthreads();

        v16bf afrag[4], bfrag[4];
#pragma unroll
        for (int mi = 0; mi < 4; ++mi)
            afrag[mi] = load_frag(&Atile[(wm + mi * 16) * LDT], lane);
#pragma unroll
        for (int ni = 0; ni < 4; ++ni)
            bfrag[ni] = load_frag(&Btile[(wn + ni * 16) * LDT], lane);

#pragma unroll
        for (int mi = 0; mi < 4; ++mi)
#pragma unroll
            for (int ni = 0; ni < 4; ++ni)
                acc[mi][ni] = __builtin_amdgcn_wmma_f32_16x16x32_bf16(
                    false, afrag[mi], false, bfrag[ni],
                    (short)0, acc[mi][ni], false, false);
        __syncthreads();
    }

    // ---- epilogue ----
    const int lrow = (lane >> 4) * 8;   // 0 or 8
    const int lcol = lane & 15;
#pragma unroll
    for (int mi = 0; mi < 4; ++mi) {
#pragma unroll
        for (int r = 0; r < 8; ++r) {
            const int gm = m0 + wm + mi * 16 + lrow + r;  // row within expert seg
            if (gm < n_e) {
                if constexpr (!PASS2) {
                    __bf16* hrow = Hout + (size_t)(off + gm) * HID;
#pragma unroll
                    for (int ni = 0; ni < 4; ++ni) {
                        const int n = n0 + wn + ni * 16 + lcol;
                        const float v = acc[mi][ni][r] + bias[e * NDIM + n];
                        hrow[n] = (__bf16)fmaxf(v, 0.0f);
                    }
                } else {
                    float* orow = O + (size_t)idxs[off + gm] * OUTD;
#pragma unroll
                    for (int ni = 0; ni < 4; ++ni) {
                        const int n = n0 + wn + ni * 16 + lcol;
                        orow[n] = acc[mi][ni][r] + bias[e * NDIM + n];
                    }
                }
            }
        }
    }
}

// ------------------------------------------------------------- normalize ----
// One wave per 256-wide chunk; in-place x / ||x||_2.
__global__ __launch_bounds__(256) void norm_kernel(float* __restrict__ O)
{
    const int wave = threadIdx.x >> 5;
    const int lane = threadIdx.x & 31;
    float* p = O + ((size_t)blockIdx.x * 8 + wave) * 256;

    float v[8];
    float s = 0.0f;
#pragma unroll
    for (int i = 0; i < 8; ++i) {
        v[i] = p[lane + i * 32];
        s += v[i] * v[i];
    }
#pragma unroll
    for (int d = 16; d >= 1; d >>= 1) s += __shfl_xor(s, d, 32);
    const float inv = 1.0f / sqrtf(s);
#pragma unroll
    for (int i = 0; i < 8; ++i) p[lane + i * 32] = v[i] * inv;
}

// ------------------------------------------------------------------ launch ---
extern "C" void kernel_launch(void* const* d_in, const int* in_sizes, int n_in,
                              void* d_out, int out_size, void* d_ws, size_t ws_size,
                              hipStream_t stream)
{
    const float* x  = (const float*)d_in[0];
    const float* W1 = (const float*)d_in[1];
    const float* b1 = (const float*)d_in[2];
    const float* W2 = (const float*)d_in[3];
    const float* b2 = (const float*)d_in[4];
    float* out = (float*)d_out;

    // Workspace layout: [0,64)B segoff(6 ints) | [64,4160)B idx[1024] | [8192,..) H bf16
    int*    segoff = (int*)d_ws;
    int*    idxs   = segoff + 16;
    __bf16* H      = (__bf16*)((char*)d_ws + 8192);   // 1024*2048 bf16 = 4 MB

    route_kernel<<<1, 32, 0, stream>>>(x, segoff, idxs);

    // Pass 1: x -> H (relu), per-expert grouped GEMM. Grid: N tiles x M tiles x experts.
    moe_gemm<false><<<dim3(HID / 128, B_TOK / 256, NEXP), 256, 0, stream>>>(
        x, W1, b1, nullptr, H, nullptr, segoff, idxs);

    // Pass 2: H -> out (unnormalized), scatter by token id.
    moe_gemm<true><<<dim3(OUTD / 128, B_TOK / 256, NEXP), 256, 0, stream>>>(
        nullptr, W2, b2, H, nullptr, out, segoff, idxs);

    // Pass 3: in-place L2 normalize each 256-chunk (B*32 chunks, 8 per block).
    norm_kernel<<<(B_TOK * 32) / 8, 256, 0, stream>>>(out);
}
